// GraphConv_21818433864287
// MI455X (gfx1250) — matrix operand order repaired and checked by reading, not compiled, not run on hardware
//
#include <hip/hip_runtime.h>
#include <hip/hip_bf16.h>
#include <cstdint>

#define N_ATOMS 100000
#define M_NBR   12
#define A_DIM   128
#define B_DIM   64
#define K_DIM   320          // 2A + B
#define O_DIM   256          // 2A
#define R_ROWS  (N_ATOMS * M_NBR)   // 1,200,000 rows (divisible by 16)
#define BN_EPS  1e-5f

typedef __attribute__((ext_vector_type(16))) __bf16 v16bf;
typedef __attribute__((ext_vector_type(8)))  __bf16 v8bf;
typedef __attribute__((ext_vector_type(4)))  __bf16 v4bf;
typedef __attribute__((ext_vector_type(8)))  float  v8f;

// packed f32x4 -> bf16x4 convert + single 8B LDS store
__device__ __forceinline__ void gc_store4(__bf16* p, float4 v) {
    v4bf r;
    r[0] = (__bf16)v.x; r[1] = (__bf16)v.y;
    r[2] = (__bf16)v.z; r[3] = (__bf16)v.w;
    *(v4bf*)p = r;
}

// ---------------------------------------------------------------------------
// Kernel 0: W [256,320] f32 (torch [out,in]) -> Wt [256][320] bf16 (N-major, K contiguous)
// ---------------------------------------------------------------------------
__global__ void gc_wconv(const float* __restrict__ W, __bf16* __restrict__ Wt) {
    int idx = blockIdx.x * blockDim.x + threadIdx.x;
    if (idx < O_DIM * K_DIM) {
        int o = idx / K_DIM;
        int k = idx - o * K_DIM;
        Wt[o * K_DIM + k] = (__bf16)W[idx];
    }
}

// ---------------------------------------------------------------------------
// Kernel 1: fused gather + bf16 WMMA GEMM + bias + per-channel stats
// Block: 256 thr = 8 waves; computes rows [16*bid, 16*bid+16) x 256 cols.
// ---------------------------------------------------------------------------
__global__ __launch_bounds__(256) void gc_gemm(
    const float* __restrict__ atom, const float* __restrict__ bond,
    const int* __restrict__ bond_idx, const __bf16* __restrict__ Wt,
    const float* __restrict__ bias, float* __restrict__ G,
    float* __restrict__ gSum, float* __restrict__ gSq)
{
    __shared__ __bf16 X[16][328];          // 16 rows x 320 (row stride 656B, 8B-aligned)
    __shared__ float  bSum[O_DIM];
    __shared__ float  bSq[O_DIM];

    const int tid = threadIdx.x;
    bSum[tid] = 0.0f;
    bSq[tid]  = 0.0f;

    // ---- stage 16 gathered rows into LDS as bf16 (16 threads per row) ----
    // Each thread: 5 coalesced float4 loads, 5 packed-bf16 8B LDS stores.
    const int rowBase = blockIdx.x * 16;
    {
        const int i   = tid >> 4;          // row in tile
        const int sub = tid & 15;
        const int r   = rowBase + i;
        const int n   = r / M_NBR;
        const int m   = r - n * M_NBR;
        const int nbr = bond_idx[n * M_NBR + m];
        const float4* selfp = (const float4*)(atom + (size_t)n   * A_DIM);
        const float4* nbrp  = (const float4*)(atom + (size_t)nbr * A_DIM);
        const float4* bondp = (const float4*)(bond + ((size_t)n * M_NBR + m) * B_DIM);
        gc_store4(&X[i][4 * sub],              selfp[sub]);        // atom_self lo
        gc_store4(&X[i][4 * (sub + 16)],       selfp[sub + 16]);   // atom_self hi
        gc_store4(&X[i][128 + 4 * sub],        nbrp[sub]);         // atom_nbr lo
        gc_store4(&X[i][128 + 4 * (sub + 16)], nbrp[sub + 16]);    // atom_nbr hi
        gc_store4(&X[i][256 + 4 * sub],        bondp[sub]);        // bond
    }
    __syncthreads();

    // ---- WMMA: wave w owns column tiles 2w and 2w+1 ----
    const int wave = tid >> 5;
    const int lane = tid & 31;
    const int arow = lane & 15;            // A: row = lane%16
    const int koff = (lane >> 4) << 3;     // A: K-half split at lane 16
    const int n0 = (wave * 2) * 16;
    const int n1 = n0 + 16;

    v8f c0 = {};
    v8f c1 = {};
    #pragma unroll
    for (int kk = 0; kk < K_DIM; kk += 32) {
        union { v16bf v; v8bf h[2]; } a;
        a.h[0] = *(const v8bf*)&X[arow][kk + koff];        // K = kk+koff .. +7
        a.h[1] = *(const v8bf*)&X[arow][kk + 16 + koff];   // K = kk+16+koff .. +7
        const __bf16* w0 = Wt + (size_t)(n0 + arow) * K_DIM + kk + koff; // B: col = lane%16
        const __bf16* w1 = Wt + (size_t)(n1 + arow) * K_DIM + kk + koff;
        union { v16bf v; v8bf h[2]; } bb0, bb1;
        bb0.h[0] = *(const v8bf*)w0; bb0.h[1] = *(const v8bf*)(w0 + 16);
        bb1.h[0] = *(const v8bf*)w1; bb1.h[1] = *(const v8bf*)(w1 + 16);
        v16bf b0 = bb0.v;
        v16bf b1 = bb1.v;
        c0 = __builtin_amdgcn_wmma_f32_16x16x32_bf16(false, a.v, false, b0,
                                                     (short)0, c0, false, false);
        c1 = __builtin_amdgcn_wmma_f32_16x16x32_bf16(false, a.v, false, b1,
                                                     (short)0, c1, false, false);
    }

    // ---- epilogue: bias, store, per-channel partial stats ----
    const int crowHalf = (lane >> 4) * 8;  // C/D: lanes 16-31 hold rows 8..15
    const int ccol     = lane & 15;
    const float bias0 = bias[n0 + ccol];
    const float bias1 = bias[n1 + ccol];
    float s0 = 0.f, q0 = 0.f, s1 = 0.f, q1 = 0.f;
    #pragma unroll
    for (int j = 0; j < 8; ++j) {
        const size_t gr = (size_t)(rowBase + j + crowHalf) * O_DIM;
        float v0 = c0[j] + bias0;
        float v1 = c1[j] + bias1;
        G[gr + n0 + ccol] = v0;
        G[gr + n1 + ccol] = v1;
        s0 += v0; q0 += v0 * v0;
        s1 += v1; q1 += v1 * v1;
    }
    atomicAdd(&bSum[n0 + ccol], s0);
    atomicAdd(&bSq [n0 + ccol], q0);
    atomicAdd(&bSum[n1 + ccol], s1);
    atomicAdd(&bSq [n1 + ccol], q1);
    __syncthreads();
    atomicAdd(&gSum[tid], bSum[tid]);      // one global atomic / channel / block
    atomicAdd(&gSq [tid], bSq [tid]);
}

// ---------------------------------------------------------------------------
// Kernel 2/4: finalize batch-norm stats (biased variance, eps=1e-5)
// ---------------------------------------------------------------------------
__global__ void gc_finalize(const float* __restrict__ sum, const float* __restrict__ sq,
                            float* __restrict__ mean, float* __restrict__ inv,
                            int C, float invR)
{
    int c = blockIdx.x * blockDim.x + threadIdx.x;
    if (c < C) {
        float mu  = sum[c] * invR;
        float var = fmaxf(sq[c] * invR - mu * mu, 0.0f);
        mean[c] = mu;
        inv[c]  = rsqrtf(var + BN_EPS);
    }
}

__device__ __forceinline__ float gc_softplus(float x) {
    return fmaxf(x, 0.0f) + log1pf(expf(-fabsf(x)));
}

// ---------------------------------------------------------------------------
// Kernel 3: BN1 -> sigmoid(filter)*softplus(core), sum over M, + BN2 stats
// Thread owns channel a = tid&127 and a strided set of n.
// ---------------------------------------------------------------------------
__global__ __launch_bounds__(256) void gc_pass2(
    const float* __restrict__ G, const float* __restrict__ mean1,
    const float* __restrict__ inv1, float* __restrict__ summed,
    float* __restrict__ gSum2, float* __restrict__ gSq2)
{
    const int a = threadIdx.x & 127;
    const int slot = blockIdx.x * 2 + (threadIdx.x >> 7);
    const int stride = gridDim.x * 2;
    const float mF = mean1[a],       iF = inv1[a];
    const float mC = mean1[a + 128], iC = inv1[a + 128];
    float ls = 0.f, lq = 0.f;
    for (int n = slot; n < N_ATOMS; n += stride) {
        const float* gp = G + (size_t)n * M_NBR * O_DIM + a;
        float acc = 0.f;
        #pragma unroll
        for (int m = 0; m < M_NBR; ++m) {
            float f  = (gp[m * O_DIM]       - mF) * iF;
            float cg = (gp[m * O_DIM + 128] - mC) * iC;
            float sig = 1.0f / (1.0f + expf(-f));
            acc += sig * gc_softplus(cg);
        }
        summed[(size_t)n * A_DIM + a] = acc;
        ls += acc; lq += acc * acc;
    }
    atomicAdd(&gSum2[a], ls);
    atomicAdd(&gSq2[a], lq);
}

// ---------------------------------------------------------------------------
// Kernel 5: out = softplus(atom + BN2(summed))
// ---------------------------------------------------------------------------
__global__ void gc_final(const float* __restrict__ atom, const float* __restrict__ summed,
                         const float* __restrict__ mean2, const float* __restrict__ inv2,
                         float* __restrict__ out)
{
    int idx = blockIdx.x * blockDim.x + threadIdx.x;
    if (idx < N_ATOMS * A_DIM) {
        int a = idx & 127;
        float s = (summed[idx] - mean2[a]) * inv2[a];
        out[idx] = gc_softplus(atom[idx] + s);
    }
}

// ---------------------------------------------------------------------------
extern "C" void kernel_launch(void* const* d_in, const int* in_sizes, int n_in,
                              void* d_out, int out_size, void* d_ws, size_t ws_size,
                              hipStream_t stream)
{
    const float* atom = (const float*)d_in[0];
    const float* bond = (const float*)d_in[1];
    const int*   bidx = (const int*)d_in[2];   // JAX w/o x64 -> int32
    const float* W    = (const float*)d_in[3];
    const float* bias = (const float*)d_in[4];
    float* out = (float*)d_out;

    // workspace layout (all offsets 256B-aligned)
    char*   ws   = (char*)d_ws;
    float*  G    = (float*)ws;                                   // 1.229 GB
    size_t  off  = (size_t)R_ROWS * O_DIM * sizeof(float);
    __bf16* Wt   = (__bf16*)(ws + off);                          // 160 KB
    off += (size_t)K_DIM * O_DIM * sizeof(__bf16);
    float* summed = (float*)(ws + off);                          // 51.2 MB
    off += (size_t)N_ATOMS * A_DIM * sizeof(float);
    float* stats = (float*)(ws + off);                           // 6 KB
    float* gSum1 = stats;        float* gSq1 = stats + 256;
    float* mean1 = stats + 512;  float* inv1 = stats + 768;
    float* gSum2 = stats + 1024; float* gSq2 = stats + 1152;
    float* mean2 = stats + 1280; float* inv2 = stats + 1408;

    hipMemsetAsync(stats, 0, 1536 * sizeof(float), stream);      // fresh accumulators

    gc_wconv<<<(O_DIM * K_DIM + 255) / 256, 256, 0, stream>>>(W, Wt);
    gc_gemm<<<R_ROWS / 16, 256, 0, stream>>>(atom, bond, bidx, Wt, bias, G, gSum1, gSq1);
    gc_finalize<<<1, 256, 0, stream>>>(gSum1, gSq1, mean1, inv1, O_DIM, 1.0f / (float)R_ROWS);
    gc_pass2<<<1024, 256, 0, stream>>>(G, mean1, inv1, summed, gSum2, gSq2);
    gc_finalize<<<1, 128, 0, stream>>>(gSum2, gSq2, mean2, inv2, A_DIM, 1.0f / (float)N_ATOMS);
    gc_final<<<(N_ATOMS * A_DIM + 255) / 256, 256, 0, stream>>>(atom, summed, mean2, inv2, out);
}
